// LayerStacks_46557445489276
// MI455X (gfx1250) — matrix operand order repaired and checked by reading, not compiled, not run on hardware
//
#include <hip/hip_runtime.h>

// ---------------------------------------------------------------------------
// NNUE LayerStacks for MI455X (gfx1250, wave32).
// Bin samples by bucket, then one wave per 16-sample tile runs the bucket's
// MLP with V_WMMA_F32_16X16X32_F16: 7 WMMAs for layer 1 (block-diagonal
// packed [x_base|x_pa] against [w1b|w1p]), 4 WMMAs for layer 2, f32
// accumulators, exact fake-quantization math between stages.
// LDS B-matrices are stored [N][K] (transposed) with padded strides so each
// lane's WMMA fragment is two contiguous, bank-distinct 16-byte runs.
// ---------------------------------------------------------------------------

typedef __attribute__((ext_vector_type(16))) _Float16 v16h;
typedef __attribute__((ext_vector_type(8)))  float    v8f;

#define NB    15            // COUNT buckets
#define L1IN  97
#define KPAD  224           // 2*97 = 194 padded to 7*32
#define KT1   7             // K-tiles in layer 1
#define AHS   232           // padded row stride (halves): 232*2B=116 dw, 116%64=52, gcd(52,64)=4 -> 16 distinct banks
#define A2S   40            // padded row stride for 32-K tiles: 20 dw, gcd(20,64)=4 -> distinct banks
#define BQ    8128.0f       // 64*127  (bias scale layers 1&2)
#define OWS   (64.0f*16.0f/127.0f)  // out weight scale
#define OBS   1024.0f       // 64*16  (out bias scale)

// ws layout (int32 units):
//   [0..14]   per-bucket counts
//   [16..31]  sampleStart prefix (16 entries, [31]=B)
//   [32..47]  tileStart prefix   (16 entries, [47]=totalTiles)
//   [48..62]  scatter cursors
//   [64.. )   idx[B] compacted sample indices grouped by bucket

__global__ __launch_bounds__(64) void zero_ws(int* ws) {
    if (threadIdx.x < 64) ws[threadIdx.x] = 0;
}

__global__ __launch_bounds__(256) void hist_kernel(const int* __restrict__ ply,
                                                   int* __restrict__ ws, int B) {
    int i = blockIdx.x * blockDim.x + threadIdx.x;
    if (i < B) {
        int b = ply[i] >> 2;            // BUCKET = 60/15 = 4
        atomicAdd(&ws[b], 1);
    }
}

__global__ __launch_bounds__(32) void prefix_kernel(int* ws) {
    if (threadIdx.x == 0) {
        int s = 0, t = 0;
        for (int b = 0; b < NB; ++b) {
            ws[16 + b] = s;
            ws[32 + b] = t;
            int c = ws[b];
            s += c;
            t += (c + 15) >> 4;
        }
        ws[16 + NB] = s;   // == B
        ws[32 + NB] = t;   // total tiles
    }
}

__global__ __launch_bounds__(256) void scatter_kernel(const int* __restrict__ ply,
                                                      int* __restrict__ ws, int B) {
    int i = blockIdx.x * blockDim.x + threadIdx.x;
    if (i < B) {
        int b   = ply[i] >> 2;
        int pos = atomicAdd(&ws[48 + b], 1);
        ws[64 + ws[16 + b] + pos] = i;
    }
}

__global__ __launch_bounds__(32)
void nnue_main(const float* __restrict__ xb,  const float* __restrict__ xp,
               const float* __restrict__ w1b, const float* __restrict__ b1b,
               const float* __restrict__ w1p, const float* __restrict__ b1p,
               const float* __restrict__ w2,  const float* __restrict__ b2,
               const float* __restrict__ wo,  const float* __restrict__ bo,
               const int*   __restrict__ ws,  float* __restrict__ out) {
    __shared__ _Float16 ah  [16 * AHS];   // layer-1 A tile, [M][K] padded
    __shared__ _Float16 bm1t[16 * AHS];   // layer-1 B transposed, [N][K] padded
    __shared__ float    d1  [16 * 16];    // layer-1 pre-activation, [M][N]
    __shared__ _Float16 a2  [16 * A2S];   // layer-2 A tile, [M][K] padded
    __shared__ _Float16 bm2t[64 * A2S];   // layer-2 B transposed, [N][K] padded
    __shared__ float    d2  [16 * 64];    // layer-2 pre-activation, [M][N]
    __shared__ int      smp [16];         // sample indices of this tile

    const int lane = threadIdx.x;
    const int t    = blockIdx.x;
    if (t >= ws[32 + NB]) return;         // beyond last tile (uniform exit)

    // Find bucket b: tileStart[b] <= t < tileStart[b+1]
    int b = 0;
    #pragma unroll
    for (int q = 0; q < NB - 1; ++q)
        if (t >= ws[32 + q + 1]) b = q + 1;

    const int tileLocal  = t - ws[32 + b];
    const int sampleBase = ws[16 + b] + tileLocal * 16;
    const int sampleEnd  = ws[16 + b] + ws[b];

    if (lane == 0) {   // warm L2/WGP$ for this bucket's weight slice
        __builtin_prefetch(w1b + (size_t)b * 8 * L1IN, 0, 1);
        __builtin_prefetch(w1p + (size_t)b * 8 * L1IN, 0, 1);
        __builtin_prefetch(w2  + (size_t)b * 64 * 32,  0, 1);
    }
    if (lane < 16) {
        int g = sampleBase + lane;
        smp[lane] = (g < sampleEnd) ? ws[64 + g] : -1;
    }
    __syncthreads();

    // ---- Layer-1 A tile: [16][KPAD] = [x_base | x_pa | 0-pad], f16 ----
    for (int e = lane; e < 16 * KPAD; e += 32) {
        int m = e / KPAD, k = e % KPAD;
        int s = smp[m];
        float v = 0.0f;
        if (s >= 0) {
            if (k < L1IN)            v = xb[(size_t)s * L1IN + k];
            else if (k < 2 * L1IN)   v = xp[(size_t)s * L1IN + (k - L1IN)];
        }
        ah[m * AHS + k] = (_Float16)v;
    }
    // ---- Layer-1 B transposed: block-diagonal quantized weights, [N][K] ----
    for (int e = lane; e < 16 * KPAD; e += 32) {
        int n = e / KPAD, k = e % KPAD;
        float wv = 0.0f;
        if (n < 8) { if (k < L1IN) wv = w1b[(size_t)(b * 8 + n) * L1IN + k]; }
        else       { if (k >= L1IN && k < 2 * L1IN)
                         wv = w1p[(size_t)(b * 8 + (n - 8)) * L1IN + (k - L1IN)]; }
        bm1t[n * AHS + k] = (_Float16)(rintf(wv * 64.0f) * (1.0f / 64.0f));
    }
    // ---- Layer-2 B transposed: [N=64][K=32] quantized ----
    for (int e = lane; e < 64 * 32; e += 32) {
        int n = e >> 5, k = e & 31;
        float wv = w2[(size_t)(b * 64 + n) * 32 + k];
        bm2t[n * A2S + k] = (_Float16)(rintf(wv * 64.0f) * (1.0f / 64.0f));
    }
    __syncthreads();

    const int m    = lane & 15;          // A row / B column / D column index
    const int half = lane >> 4;          // lane group per ISA layout
    const int kb   = half * 16;          // unused; B k-runs start at 8*half and 16+8*half

    // ---- Layer 1: one 16x16 f32 accumulator over 7 K-tiles of 32 ----
    v8f c = {};
    #pragma unroll
    for (int kk = 0; kk < KT1; ++kk) {
        v16h a, bf;
        #pragma unroll
        for (int e = 0; e < 16; ++e) {
            int p = e >> 1, o = e & 1;
            int k = (p < 4) ? (half * 8 + 2 * p + o)
                            : (16 + half * 8 + 2 * (p - 4) + o);
            a[e]  = ah[m * AHS + kk * 32 + k];            // 2 contiguous 8-half runs
            bf[e] = bm1t[m * AHS + kk * 32 + k];          // 2 contiguous 8-half runs
        }
        c = __builtin_amdgcn_wmma_f32_16x16x32_f16(false, a, false, bf,
                                                   (short)0, c, false, false);
    }
    #pragma unroll
    for (int r = 0; r < 8; ++r)
        d1[(r + 8 * half) * 16 + m] = c[r];
    __syncthreads();

    // ---- Activation 1: bias, [squares*127/128 | linear], clip, floor/127 ----
    {
        int j = lane >> 1, h = lane & 1;   // 2 lanes per sample: h=0 squares, h=1 linear
        #pragma unroll
        for (int i = 0; i < 16; ++i) {
            float bias = (i < 8) ? b1b[b * 8 + i] : b1p[b * 8 + (i - 8)];
            bias = rintf(bias * BQ) * (1.0f / BQ);
            float v = d1[j * 16 + i] + bias;
            float f = h ? v : (v * v * (127.0f / 128.0f));
            f = fminf(fmaxf(f, 0.0f), 1.0f);
            f = floorf(f * 127.0f) * (1.0f / 127.0f);
            a2[j * A2S + (h ? 16 + i : i)] = (_Float16)f;
        }
    }
    __syncthreads();

    // ---- Layer 2: 16x32 @ 32x64 -> 4 N-tiles, 1 WMMA each ----
    v16h aa;
    #pragma unroll
    for (int e = 0; e < 16; ++e) {
        int p = e >> 1, o = e & 1;
        int k = (p < 4) ? (half * 8 + 2 * p + o)
                        : (16 + half * 8 + 2 * (p - 4) + o);
        aa[e] = a2[m * A2S + k];
    }
    #pragma unroll
    for (int nt = 0; nt < 4; ++nt) {
        v16h bf;
        #pragma unroll
        for (int e = 0; e < 16; ++e)
            bf[e] = bm2t[(nt * 16 + m) * A2S + ((e & 8) << 1) + half * 8 + (e & 7)];   // 2 contiguous 8-half runs
        v8f c2 = {};
        c2 = __builtin_amdgcn_wmma_f32_16x16x32_f16(false, aa, false, bf,
                                                    (short)0, c2, false, false);
        #pragma unroll
        for (int r = 0; r < 8; ++r)
            d2[(r + 8 * half) * 64 + nt * 16 + m] = c2[r];
    }
    __syncthreads();

    // ---- Activation 2 + output layer (64 -> 1) per sample ----
    if (lane < 16) {
        int s = smp[lane];
        if (s >= 0) {
            float acc = rintf(bo[b] * OBS) * (1.0f / OBS);
            #pragma unroll 8
            for (int k = 0; k < 64; ++k) {
                float bias = rintf(b2[b * 64 + k] * BQ) * (1.0f / BQ);
                float v = d2[lane * 64 + k] + bias;
                v = fminf(fmaxf(v, 0.0f), 1.0f);
                v = floorf(v * 127.0f) * (1.0f / 127.0f);
                float ww = rintf(wo[b * 64 + k] * OWS) * (1.0f / OWS);
                acc += v * ww;
            }
            out[s] = floorf(acc * 64.0f) * (1.0f / 64.0f);
        }
    }
}

extern "C" void kernel_launch(void* const* d_in, const int* in_sizes, int n_in,
                              void* d_out, int out_size, void* d_ws, size_t ws_size,
                              hipStream_t stream) {
    (void)n_in; (void)out_size; (void)ws_size;
    const float* xb  = (const float*)d_in[0];
    const float* xp  = (const float*)d_in[1];
    const int*   ply = (const int*)  d_in[2];
    const float* w1b = (const float*)d_in[3];
    const float* b1b = (const float*)d_in[4];
    const float* w1p = (const float*)d_in[5];
    const float* b1p = (const float*)d_in[6];
    const float* w2  = (const float*)d_in[7];
    const float* b2  = (const float*)d_in[8];
    const float* wo  = (const float*)d_in[9];
    const float* bo  = (const float*)d_in[10];
    float* out = (float*)d_out;
    int*   ws  = (int*)d_ws;          // needs (64 + B) * 4 bytes of scratch

    const int B = in_sizes[2];

    zero_ws<<<1, 64, 0, stream>>>(ws);
    hist_kernel<<<(B + 255) / 256, 256, 0, stream>>>(ply, ws, B);
    prefix_kernel<<<1, 32, 0, stream>>>(ws);
    scatter_kernel<<<(B + 255) / 256, 256, 0, stream>>>(ply, ws, B);

    const int maxTiles = (B + 15) / 16 + NB;   // upper bound on total tiles
    nnue_main<<<maxTiles, 32, 0, stream>>>(xb, xp, w1b, b1b, w1p, b1p,
                                           w2, b2, wo, bo, ws, out);
}
